// PatchLossOptimizedV4_43052752175789
// MI455X (gfx1250) — matrix operand order repaired and checked by reading, not compiled, not run on hardware
//
#include <hip/hip_runtime.h>
#include <hip/hip_fp16.h>

typedef _Float16 v16h __attribute__((ext_vector_type(16)));
typedef _Float16 v8h  __attribute__((ext_vector_type(8)));
typedef float    v8f  __attribute__((ext_vector_type(8)));

#define B_  16
#define T_  1024
#define D_  256

// ---------------- init: zero output scalar + precompute 1/(2*sigma^2) ----------------
__global__ void init_kernel(const float* __restrict__ sigma, float* __restrict__ rws,
                            float* __restrict__ out) {
  if (threadIdx.x == 0) out[0] = 0.0f;
  if (threadIdx.x < 4) {
    const float s = sigma[threadIdx.x];
    rws[threadIdx.x] = 1.0f / (2.0f * s * s);
  }
}

// ---------------- convert z (f32) -> zh (f16), row-major [B,T,D] ----------------
__global__ void cvt_kernel(const float* __restrict__ z, _Float16* __restrict__ zh, int n) {
  int i = blockIdx.x * blockDim.x + threadIdx.x;
  if (i < n) zh[i] = (_Float16)z[i];
}

// ---------------- z2[b,t] = sum_d z[b,t,d]^2 ; one wave32 per row ----------------
__global__ __launch_bounds__(256) void rowsq_kernel(const float* __restrict__ z,
                                                    float* __restrict__ z2) {
  const int wid  = threadIdx.x >> 5;
  const int lane = threadIdx.x & 31;
  const int row  = blockIdx.x * 8 + wid;          // row in [0, B*T)
  const float* p = z + (size_t)row * D_;
  float s = 0.f;
#pragma unroll
  for (int i = 0; i < D_ / 32; ++i) { float v = p[lane + i * 32]; s += v * v; }
#pragma unroll
  for (int off = 16; off > 0; off >>= 1) s += __shfl_down(s, off, 32);
  if (lane == 0) z2[row] = s;
}

// ---------------- fused: G tile via WMMA + w = exp(...) + scalar reduction ----------------
// One wave per 16x16 (t,s) tile. Block = 8 waves covering 16(t) x 128(s).
__global__ __launch_bounds__(256) void fused_loss_kernel(
    const _Float16* __restrict__ zh,   // [B,T,D] f16
    const float*    __restrict__ z2,   // [B,T]
    const float4*   __restrict__ g4,   // gt_dT viewed as float4 [B,T,T]
    const float*    __restrict__ rws,  // [4] = 1/(2*sigma^2)
    float*          __restrict__ out)
{
  const int lane = threadIdx.x & 31;
  const int wid  = threadIdx.x >> 5;
  const int b    = blockIdx.z;
  const int t0   = blockIdx.y * 16;
  const int s0   = blockIdx.x * 128 + wid * 16;

  const int hg = lane >> 4;     // lane half-group (0/1): selects K-half per ISA layout
  const int lr = lane & 15;     // row-in-tile for A/B fragments; N for C/D

  // precomputed reciprocal channel scales (L2-resident, 16B)
  const float r0 = rws[0], r1 = rws[1], r2 = rws[2], r3 = rws[3];

  // Issue the dominant HBM stream (gt_dT tile, 4KB/wave) FIRST so its latency
  // hides under the WMMA loop + L2-resident zh loads.
  // C/D layout: lane holds G[m, n] with n = lr, m = v + 8*hg for v = 0..7
  const float4* gp = g4 + ((size_t)b * T_ + (size_t)(t0 + 8 * hg)) * T_ + (s0 + lr);
  float4 g[8];
#pragma unroll
  for (int v = 0; v < 8; ++v) g[v] = gp[(size_t)v * T_];

  // A-fragment rows (t-block) and B-fragment rows (s-block); B = Z_s^T so
  // column n of B is row s0+n of Z -> same row-major staging as A.
  const _Float16* za = zh + ((size_t)b * T_ + (size_t)(t0 + lr)) * D_;
  const _Float16* zb = zh + ((size_t)b * T_ + (size_t)(s0 + lr)) * D_;

  v8f c = {};
#pragma unroll
  for (int k0 = 0; k0 < D_; k0 += 32) {
    // A 16x32 f16 layout: lane group 0 holds K {k0..k0+7, k0+16..k0+23},
    // lane group 1 holds K {k0+8..k0+15, k0+24..k0+31}
    union { v16h v; v8h h[2]; } a;
    a.h[0] = *(const v8h*)(za + k0 + 8 * hg);
    a.h[1] = *(const v8h*)(za + k0 + 16 + 8 * hg);
    // B 32x16 f16 layout: lane holds column N=lr, same K order as A
    union { v16h v; v8h h[2]; } bm;
    bm.h[0] = *(const v8h*)(zb + k0 + 8 * hg);
    bm.h[1] = *(const v8h*)(zb + k0 + 16 + 8 * hg);
    v16h bmat = bm.v;
    c = __builtin_amdgcn_wmma_f32_16x16x32_f16(false, a.v, false, bmat,
                                               (short)0, c, false, false);
  }

  const float* z2b = z2 + (size_t)b * T_;

  float acc = 0.f;
#pragma unroll
  for (int v = 0; v < 8; ++v) {
    const int t = t0 + 8 * hg + v;
    const float e = g[v].x * g[v].x * r0 + g[v].y * g[v].y * r1 +
                    g[v].z * g[v].z * r2 + g[v].w * g[v].w * r3;
    const float w = __expf(e);          // faithful: exponent NOT negated
    acc += w * (z2b[t] - c[v]);
  }

  // wave32 reduction, then one f32 atomic per wave
#pragma unroll
  for (int off = 16; off > 0; off >>= 1) acc += __shfl_down(acc, off, 32);
  if (lane == 0) {
    const float scale = 2.0f / ((float)B_ * (float)T_ * (float)T_);
    atomicAdd(out, acc * scale);
  }
}

extern "C" void kernel_launch(void* const* d_in, const int* in_sizes, int n_in,
                              void* d_out, int out_size, void* d_ws, size_t ws_size,
                              hipStream_t stream) {
  const float* z     = (const float*)d_in[0];   // [B,T,D]
  const float* gt    = (const float*)d_in[1];   // [B,T,T,4]
  const float* sigma = (const float*)d_in[2];   // [4]
  float* out = (float*)d_out;

  _Float16* zh  = (_Float16*)d_ws;                                   // 8.39 MB
  float*    z2  = (float*)((char*)d_ws + (size_t)B_ * T_ * D_ * 2);  // 64 KB
  float*    rws = z2 + (size_t)B_ * T_;                              // 16 B

  init_kernel<<<1, 32, 0, stream>>>(sigma, rws, out);

  const int n = B_ * T_ * D_;
  cvt_kernel<<<(n + 255) / 256, 256, 0, stream>>>(z, zh, n);

  rowsq_kernel<<<(B_ * T_) / 8, 256, 0, stream>>>(z, z2);

  dim3 grid(T_ / 128, T_ / 16, B_);   // (8, 64, 16)
  fused_loss_kernel<<<grid, 256, 0, stream>>>(zh, z2, (const float4*)gt, rws, out);
}
